// GNN_learner_9809705304350
// MI455X (gfx1250) — matrix operand order repaired and checked by reading, not compiled, not run on hardware
//
#include <hip/hip_runtime.h>
#include <hip/hip_bf16.h>

// ---------------------------------------------------------------------------
// Types
// ---------------------------------------------------------------------------
typedef __attribute__((ext_vector_type(16))) __bf16 v16bf;
typedef __attribute__((ext_vector_type(8)))  __bf16 v8bf;
typedef __attribute__((ext_vector_type(4)))  __bf16 v4bf;
typedef __attribute__((ext_vector_type(8)))  float  v8f;
typedef __attribute__((ext_vector_type(4)))  float  v4f;
typedef __attribute__((ext_vector_type(4)))  unsigned int v4u;
typedef int v4i __attribute__((vector_size(4 * sizeof(int))));

union bfrag { v16bf v; v8bf h[2]; };

#define GCN_N 8192
#define GCN_D 512
#define GCN_K 31

// ---- gfx1250 async global->LDS copy support (probe via __has_builtin) ------
#if __has_builtin(__builtin_amdgcn_global_load_async_to_lds_b128)
#define USE_ASYNC 1
#else
#define USE_ASYNC 0
#endif

#if __has_builtin(__builtin_amdgcn_s_wait_asynccnt)
#define WAIT_ASYNC(n) __builtin_amdgcn_s_wait_asynccnt(n)
#else
#define WAIT_ASYNC(n) asm volatile("s_wait_asynccnt %0" :: "i"(n) : "memory")
#endif

#define AS1 __attribute__((address_space(1)))
#define AS3 __attribute__((address_space(3)))
__device__ __forceinline__ AS1 v4i* as_global_v4(const void* p) {
    return (AS1 v4i*)(unsigned long long)p;
}
__device__ __forceinline__ AS3 v4i* as_lds_v4(const void* p) {
    return (AS3 v4i*)(unsigned)(unsigned long long)p;
}

// ---------------------------------------------------------------------------
// f32 -> bf16 conversion (vectorized x4, grid-stride)
// ---------------------------------------------------------------------------
__global__ void k_cvt_bf16_v4(const v4f* __restrict__ in, v4bf* __restrict__ out, long n4) {
    long i = (long)blockIdx.x * blockDim.x + threadIdx.x;
    long stride = (long)gridDim.x * blockDim.x;
    for (; i < n4; i += stride) {
        v4f x = in[i];
        v4bf o;
        o.x = (__bf16)x.x; o.y = (__bf16)x.y;
        o.z = (__bf16)x.z; o.w = (__bf16)x.w;
        out[i] = o;
    }
}

// ---------------------------------------------------------------------------
// transpose + convert (for D x D weight matrices): WT[n][k] = W[k][n]
// ---------------------------------------------------------------------------
__global__ void k_transpose_cvt(const float* __restrict__ W, __bf16* __restrict__ WT, int D) {
    int x = blockIdx.x * 32 + (threadIdx.x & 31);
    int y = blockIdx.y * 8 + (threadIdx.x >> 5);
    if (x < D && y < D)
        WT[(long)x * D + y] = (__bf16)W[(long)y * D + x];
}

// ---------------------------------------------------------------------------
// WMMA bf16 GEMM:  C[M,N] = A[M,K] * BT[N,K]^T   (both bf16 row-major)
// Block tile 128x128, K tile 64, 8 waves (wave tile 32x64), wave32.
// Async path: double-buffered LDS fed by global_load_async_to_lds_b128.
// ---------------------------------------------------------------------------
#define BM 128
#define BN 128
#define BK 64
#define LDT 72   // BK + 8 bf16 padding (144B rows, 16B aligned)

enum { EP_F32 = 0, EP_RELU_BF16 = 1, EP_TBF16 = 2 };

template<int MODE>
__global__ __launch_bounds__(256)
void k_gemm_bf16(const __bf16* __restrict__ A, const __bf16* __restrict__ BT,
                 float* __restrict__ Cf, __bf16* __restrict__ Cb,
                 int M, int N, int K, int ldc)
{
#if USE_ASYNC
    __shared__ __bf16 As[2][BM * LDT];
    __shared__ __bf16 Bs[2][BN * LDT];
#else
    __shared__ __bf16 As[1][BM * LDT];
    __shared__ __bf16 Bs[1][BN * LDT];
#endif

    const int tid  = threadIdx.x;
    const int lane = tid & 31;
    const int wid  = tid >> 5;
    const int wm0  = (wid & 3) * 32;   // wave M origin in block tile
    const int wn0  = (wid >> 2) * 64;  // wave N origin in block tile
    const long m0  = (long)blockIdx.y * BM;
    const long n0  = (long)blockIdx.x * BN;

    const int lr = lane & 15;  // 0..15
    const int lh = lane >> 4;  // 0 or 1

    // staging geometry: 1024 chunks of 8 bf16 per matrix, 4 per thread;
    // chunk = tid + 256*i  ->  row = (tid>>3) + 32*i, col8 = (tid&7)*8
    const int srow = tid >> 3;
    const int sc8  = (tid & 7) * 8;

    v8f acc[2][4];
    #pragma unroll
    for (int i = 0; i < 2; ++i)
        #pragma unroll
        for (int j = 0; j < 4; ++j) { v8f z = {}; acc[i][j] = z; }

    const int kTiles = K / BK;

    // ---- compute over one staged tile ------------------------------------
    auto compute = [&](const __bf16* __restrict__ as, const __bf16* __restrict__ bs) {
        #pragma unroll
        for (int ks = 0; ks < BK; ks += 32) {
            bfrag a[2], b[4];
            // A frag (16-bit 16x32): lanes 0-15 hold K 0-7 & 16-23,
            // lanes 16-31 hold K 8-15 & 24-31; row M = lane&15.
            #pragma unroll
            for (int mt = 0; mt < 2; ++mt) {
                int row = wm0 + mt * 16 + lr;
                int kb  = ks + lh * 8;
                a[mt].h[0] = *(const v8bf*)(&as[row * LDT + kb]);
                a[mt].h[1] = *(const v8bf*)(&as[row * LDT + kb + 16]);
            }
            // B frag (32x16): lanes 0-15 K 0-7 & 16-23, lanes 16-31 K 8-15 & 24-31;
            // column N = lane&15; Bs is N-major so K contiguous.
            #pragma unroll
            for (int nt = 0; nt < 4; ++nt) {
                int col = wn0 + nt * 16 + lr;
                int kb  = ks + lh * 8;
                b[nt].h[0] = *(const v8bf*)(&bs[col * LDT + kb]);
                b[nt].h[1] = *(const v8bf*)(&bs[col * LDT + kb + 16]);
            }
            #pragma unroll
            for (int mt = 0; mt < 2; ++mt)
                #pragma unroll
                for (int nt = 0; nt < 4; ++nt)
                    acc[mt][nt] = __builtin_amdgcn_wmma_f32_16x16x32_bf16(
                        false, a[mt].v, false, b[nt].v,
                        (short)0, acc[mt][nt], false, false);
        }
    };

#if USE_ASYNC
    // ---- async double-buffered pipeline ----------------------------------
    auto stage = [&](int kt, int buf) {
        const long k0 = (long)kt * BK;
        #pragma unroll
        for (int i = 0; i < 4; ++i) {
            int row = srow + 32 * i;
            const __bf16* ga = A  + (m0 + row) * (long)K + k0 + sc8;
            const __bf16* gb = BT + (n0 + row) * (long)K + k0 + sc8;
            __builtin_amdgcn_global_load_async_to_lds_b128(
                as_global_v4(ga), as_lds_v4(&As[buf][row * LDT + sc8]), 0, 0);
            __builtin_amdgcn_global_load_async_to_lds_b128(
                as_global_v4(gb), as_lds_v4(&Bs[buf][row * LDT + sc8]), 0, 0);
        }
    };

    stage(0, 0);
    for (int kt = 0; kt < kTiles; ++kt) {
        const int buf = kt & 1;
        if (kt + 1 < kTiles) {
            stage(kt + 1, buf ^ 1);   // stream next tile while computing this one
            WAIT_ASYNC(8);            // 8 newer ops outstanding -> current tile done
        } else {
            WAIT_ASYNC(0);
        }
        __syncthreads();
        compute(As[buf], Bs[buf]);
        __syncthreads();              // everyone done reading before buf re-staged
    }
#else
    // ---- fallback: register-staged, load-ahead ---------------------------
    v4u ra[4], rb[4];
    {
        #pragma unroll
        for (int i = 0; i < 4; ++i) {
            int row = srow + 32 * i;
            ra[i] = *(const v4u*)(A  + (m0 + row) * (long)K + sc8);
            rb[i] = *(const v4u*)(BT + (n0 + row) * (long)K + sc8);
        }
    }
    for (int kt = 0; kt < kTiles; ++kt) {
        __syncthreads();   // previous tile fully consumed
        #pragma unroll
        for (int i = 0; i < 4; ++i) {
            int row = srow + 32 * i;
            *(v4u*)(&As[0][row * LDT + sc8]) = ra[i];
            *(v4u*)(&Bs[0][row * LDT + sc8]) = rb[i];
        }
        __syncthreads();
        if (kt + 1 < kTiles) {        // issue next tile's loads before compute
            const long k0 = (long)(kt + 1) * BK;
            #pragma unroll
            for (int i = 0; i < 4; ++i) {
                int row = srow + 32 * i;
                ra[i] = *(const v4u*)(A  + (m0 + row) * (long)K + k0 + sc8);
                rb[i] = *(const v4u*)(BT + (n0 + row) * (long)K + k0 + sc8);
            }
        }
        compute(As[0], Bs[0]);
    }
#endif

    // ---- epilogue. f32 16x16 C/D: VGPR r -> M = r + 8*(lane>=16), N = lane&15.
    #pragma unroll
    for (int mt = 0; mt < 2; ++mt) {
        #pragma unroll
        for (int nt = 0; nt < 4; ++nt) {
            const long gn  = n0 + wn0 + nt * 16 + lr;
            const long gm0 = m0 + wm0 + mt * 16 + lh * 8;
            if (MODE == EP_TBF16) {
                // transposed store: lane's 8 rows are contiguous -> one 16B store
                v8bf pack;
                #pragma unroll
                for (int r = 0; r < 8; ++r) pack[r] = (__bf16)acc[mt][nt][r];
                *(v8bf*)(Cb + gn * (long)ldc + gm0) = pack;   // ldc == M
            } else {
                #pragma unroll
                for (int r = 0; r < 8; ++r) {
                    float v = acc[mt][nt][r];
                    if (MODE == EP_F32)
                        Cf[(gm0 + r) * (long)ldc + gn] = v;
                    else  // EP_RELU_BF16
                        Cb[(gm0 + r) * (long)ldc + gn] = (__bf16)fmaxf(v, 0.0f);
                }
            }
        }
    }
}

// ---------------------------------------------------------------------------
// Row L2 normalization: h2n[row] = h2[row] / max(||h2[row]||, 1e-12), bf16 out
// ---------------------------------------------------------------------------
__global__ __launch_bounds__(256)
void k_normalize(const float* __restrict__ h, __bf16* __restrict__ o, int D) {
    const int row = blockIdx.x;
    const int t = threadIdx.x;
    __shared__ float red[8];
    __shared__ float inv;
    const float* r = h + (long)row * D;
    float s = 0.0f;
    for (int i = t; i < D; i += 256) { float v = r[i]; s += v * v; }
    #pragma unroll
    for (int off = 16; off > 0; off >>= 1) s += __shfl_down(s, off, 32);
    if ((t & 31) == 0) red[t >> 5] = s;
    __syncthreads();
    if (t == 0) {
        float tot = 0.0f;
        #pragma unroll
        for (int i = 0; i < 8; ++i) tot += red[i];
        float nrm = fmaxf(sqrtf(tot), 1e-12f);
        inv = 1.0f / nrm;
    }
    __syncthreads();
    float iv = inv;
    for (int i = t; i < D; i += 256) o[(long)row * D + i] = (__bf16)(r[i] * iv);
}

// ---------------------------------------------------------------------------
// Per-row top-K mask + relu, in place on S (f32). Radix-select on
// order-preserving uint keys; one block per row, row cached in LDS.
// ---------------------------------------------------------------------------
__global__ __launch_bounds__(256)
void k_topk_relu(float* __restrict__ S, int N, int Kk) {
    __shared__ float vals[GCN_N];
    __shared__ int cnt;
    const long row = blockIdx.x;
    float* r = S + row * (long)N;
    const int t = threadIdx.x;

    for (int i = t; i < N; i += 256) vals[i] = r[i];
    __syncthreads();

    unsigned prefix = 0;
    int remaining = Kk;
    for (int bit = 31; bit >= 0; --bit) {
        if (t == 0) cnt = 0;
        __syncthreads();
        const unsigned want = (prefix >> bit) | 1u;
        int c = 0;
        for (int i = t; i < N; i += 256) {
            unsigned b = __float_as_uint(vals[i]);
            unsigned key = b ^ ((unsigned)(((int)b) >> 31) | 0x80000000u);
            if ((key >> bit) == want) ++c;
        }
        #pragma unroll
        for (int off = 16; off > 0; off >>= 1) c += __shfl_down(c, off, 32);
        if ((t & 31) == 0) atomicAdd(&cnt, c);
        __syncthreads();
        int total = cnt;
        __syncthreads();
        if (total >= remaining) prefix |= (1u << bit);
        else remaining -= total;
    }
    // keep keys >= prefix (K-th largest), relu
    for (int i = t; i < N; i += 256) {
        float v = vals[i];
        unsigned b = __float_as_uint(v);
        unsigned key = b ^ ((unsigned)(((int)b) >> 31) | 0x80000000u);
        r[i] = (key >= prefix && v > 0.0f) ? v : 0.0f;
    }
}

// ---------------------------------------------------------------------------
// Host-side launch
// ---------------------------------------------------------------------------
extern "C" void kernel_launch(void* const* d_in, const int* in_sizes, int n_in,
                              void* d_out, int out_size, void* d_ws, size_t ws_size,
                              hipStream_t stream) {
    const int N = GCN_N, D = GCN_D;
    const float* x   = (const float*)d_in[0];   // [N, D]
    const float* adj = (const float*)d_in[1];   // [N, N]
    const float* wts = (const float*)d_in[2];   // [2, D, D]
    float* out = (float*)d_out;                 // [N, N]

    // Workspace carve-up (bytes)
    char* ws = (char*)d_ws;
    size_t off = 0;
    __bf16* adjb = (__bf16*)(ws + off); off += (size_t)N * N * 2;   // 128 MB
    __bf16* xb   = (__bf16*)(ws + off); off += (size_t)N * D * 2;   //   8 MB
    __bf16* tT   = (__bf16*)(ws + off); off += (size_t)D * N * 2;   //   8 MB (D x N, transposed h@W)
    __bf16* h1b  = (__bf16*)(ws + off); off += (size_t)N * D * 2;   //   8 MB
    float*  h2f  = (float*) (ws + off); off += (size_t)N * D * 4;   //  16 MB
    __bf16* h2nb = (__bf16*)(ws + off); off += (size_t)N * D * 2;   //   8 MB
    __bf16* w0T  = (__bf16*)(ws + off); off += (size_t)D * D * 2;
    __bf16* w1T  = (__bf16*)(ws + off); off += (size_t)D * D * 2;

    // 1) Convert adj and x to bf16
    k_cvt_bf16_v4<<<8192, 256, 0, stream>>>((const v4f*)adj, (v4bf*)adjb, (long)N * N / 4);
    k_cvt_bf16_v4<<<2048, 256, 0, stream>>>((const v4f*)x,   (v4bf*)xb,   (long)N * D / 4);
    // Transposed bf16 weights
    k_transpose_cvt<<<dim3(D / 32, D / 8), 256, 0, stream>>>(wts,                 w0T, D);
    k_transpose_cvt<<<dim3(D / 32, D / 8), 256, 0, stream>>>(wts + (size_t)D * D, w1T, D);

    // 2) tT = (x @ W0)^T   [D x N bf16]
    k_gemm_bf16<EP_TBF16><<<dim3(D / BN, N / BM), 256, 0, stream>>>(
        xb, w0T, nullptr, tT, N, D, D, /*ldc=M*/N);
    // 3) h1 = relu(adj @ (x@W0))   [N x D bf16]
    k_gemm_bf16<EP_RELU_BF16><<<dim3(D / BN, N / BM), 256, 0, stream>>>(
        adjb, tT, nullptr, h1b, N, D, N, /*ldc=*/D);
    // 4) tT = (h1 @ W1)^T
    k_gemm_bf16<EP_TBF16><<<dim3(D / BN, N / BM), 256, 0, stream>>>(
        h1b, w1T, nullptr, tT, N, D, D, /*ldc=M*/N);
    // 5) h2 = adj @ (h1@W1)   [N x D f32]
    k_gemm_bf16<EP_F32><<<dim3(D / BN, N / BM), 256, 0, stream>>>(
        adjb, tT, h2f, nullptr, N, D, N, /*ldc=*/D);
    // 6) L2 normalize rows -> bf16
    k_normalize<<<N, 256, 0, stream>>>(h2f, h2nb, D);
    // 7) S = h2n @ h2n^T   [N x N f32] directly into d_out (BT == A)
    k_gemm_bf16<EP_F32><<<dim3(N / BN, N / BM), 256, 0, stream>>>(
        h2nb, h2nb, out, nullptr, N, N, D, /*ldc=*/N);
    // 8) per-row top-(K+1) sparsify + relu, in place
    k_topk_relu<<<N, 256, 0, stream>>>(out, N, GCN_K);
}
